// LightGlueSelfAttentionBlock_12146167513251
// MI455X (gfx1250) — matrix operand order repaired and checked by reading, not compiled, not run on hardware
//
#include <hip/hip_runtime.h>

// ---------------------------------------------------------------------------
// LightGlue self-attention block for MI455X (gfx1250, wave32, WMMA).
// All GEMMs + attention on V_WMMA_F32_16X16X32_BF16 (f32 accumulate).
// Attention stages K/V tiles into LDS via async global->LDS copies
// (ASYNCcnt) when the toolchain exposes the gfx1250 builtins.
// ---------------------------------------------------------------------------

typedef __bf16 bf16;
typedef __attribute__((ext_vector_type(16))) __bf16 v16bf;
typedef __attribute__((ext_vector_type(8)))  float  v8f;

union Frag { uint4 q[2]; v16bf v; };

__device__ __forceinline__ v16bf load_frag(const bf16* p0, const bf16* p1) {
  Frag f;
  f.q[0] = *(const uint4*)p0;
  f.q[1] = *(const uint4*)p1;
  return f.v;
}

// ---- async global->LDS staging (gfx1250 ASYNCcnt path) --------------------
#if defined(__has_builtin)
#if __has_builtin(__builtin_amdgcn_global_load_async_to_lds_b128) && \
    __has_builtin(__builtin_amdgcn_s_wait_asynccnt)
#define HAVE_ASYNC_LDS 1
#endif
#endif
#ifndef HAVE_ASYNC_LDS
#define HAVE_ASYNC_LDS 0
#endif

typedef int v4i __attribute__((vector_size(16)));
typedef __attribute__((address_space(1))) v4i GVec;   // global AS int4
typedef __attribute__((address_space(3))) v4i LVec;   // LDS AS int4

__device__ __forceinline__ void stage16(const bf16* g, bf16* l) {
#if HAVE_ASYNC_LDS
  __builtin_amdgcn_global_load_async_to_lds_b128((GVec*)g, (LVec*)l, 0, 0);
#else
  *(uint4*)l = *(const uint4*)g;
#endif
}

__device__ __forceinline__ void async_join() {
#if HAVE_ASYNC_LDS
  __builtin_amdgcn_s_wait_asynccnt(0);
#endif
}

// ---------------------------------------------------------------------------
// GEMM: C[M,N] = A[M,K](bf16,lda) * W[N,K]^T(bf16) + bias.
// Block tile 128x64, 256 threads = 8 waves (4M x 2N), each wave a 32x32
// register tile: 2 A-frags + 2 B-frags -> 4 WMMAs per K-step (32B/WMMA).
// EPI: 0 = bf16 store, 1 = f32 store, 2 = f32 store + residual add.
// ---------------------------------------------------------------------------
template<int EPI>
__launch_bounds__(256)
__global__ void gemm_bf16_kernel(const bf16* __restrict__ A, int lda,
                                 const bf16* __restrict__ Bw, int K,
                                 const float* __restrict__ bias,
                                 void* __restrict__ outv, int ldc, int ncol0,
                                 const float* __restrict__ resid, int ldr)
{
  const int lane = threadIdx.x & 31;
  const int w    = threadIdx.x >> 5;
  const int wm   = w & 3;
  const int wn   = w >> 2;
  const int tm   = blockIdx.x * 128;
  const int tn   = blockIdx.y * 64;
  const int hf   = lane >> 4;
  const int ln16 = lane & 15;

  const int m0 = tm + wm * 32 + ln16;
  const int m1 = m0 + 16;
  const int n0 = tn + wn * 32 + ln16;
  const int n1 = n0 + 16;

  v8f acc00 = {}, acc01 = {}, acc10 = {}, acc11 = {};

  const bf16* a0p = A  + (size_t)m0 * lda + hf * 8;
  const bf16* a1p = A  + (size_t)m1 * lda + hf * 8;
  const bf16* b0p = Bw + (size_t)n0 * K   + hf * 8;
  const bf16* b1p = Bw + (size_t)n1 * K   + hf * 8;

  for (int k0 = 0; k0 < K; k0 += 32) {
    v16bf a0 = load_frag(a0p + k0, a0p + k0 + 16);
    v16bf a1 = load_frag(a1p + k0, a1p + k0 + 16);
    v16bf b0 = load_frag(b0p + k0, b0p + k0 + 16);
    v16bf b1 = load_frag(b1p + k0, b1p + k0 + 16);
    if (k0 + 32 < K) {   // stream the next weight rows into cache
      __builtin_prefetch(b0p + k0 + 32, 0, 1);
      __builtin_prefetch(b1p + k0 + 32, 0, 1);
    }
    acc00 = __builtin_amdgcn_wmma_f32_16x16x32_bf16(false, a0, false, b0,
                                                    (short)0, acc00, false, false);
    acc01 = __builtin_amdgcn_wmma_f32_16x16x32_bf16(false, a0, false, b1,
                                                    (short)0, acc01, false, false);
    acc10 = __builtin_amdgcn_wmma_f32_16x16x32_bf16(false, a1, false, b0,
                                                    (short)0, acc10, false, false);
    acc11 = __builtin_amdgcn_wmma_f32_16x16x32_bf16(false, a1, false, b1,
                                                    (short)0, acc11, false, false);
  }

  const float bb0 = bias[n0];
  const float bb1 = bias[n1];
#pragma unroll
  for (int e = 0; e < 8; ++e) {
    const size_t r0 = (size_t)(tm + wm * 32 + hf * 8 + e);
    const size_t r1 = r0 + 16;
    float x00 = acc00[e] + bb0, x01 = acc01[e] + bb1;
    float x10 = acc10[e] + bb0, x11 = acc11[e] + bb1;
    if (EPI == 0) {
      bf16* o = (bf16*)outv;
      o[r0 * ldc + ncol0 + n0] = (bf16)x00;
      o[r0 * ldc + ncol0 + n1] = (bf16)x01;
      o[r1 * ldc + ncol0 + n0] = (bf16)x10;
      o[r1 * ldc + ncol0 + n1] = (bf16)x11;
    } else if (EPI == 1) {
      float* o = (float*)outv;
      o[r0 * ldc + ncol0 + n0] = x00;
      o[r0 * ldc + ncol0 + n1] = x01;
      o[r1 * ldc + ncol0 + n0] = x10;
      o[r1 * ldc + ncol0 + n1] = x11;
    } else {
      float* o = (float*)outv;
      o[r0 * ldc + n0] = x00 + resid[r0 * ldr + n0];
      o[r0 * ldc + n1] = x01 + resid[r0 * ldr + n1];
      o[r1 * ldc + n0] = x10 + resid[r1 * ldr + n0];
      o[r1 * ldc + n1] = x11 + resid[r1 * ldr + n1];
    }
  }
}

// ---------------------------------------------------------------------------
// Flash attention: 256 threads = 8 waves, each wave owns 16 query rows.
// kv tile = 32 rows, double-buffered in LDS via async global->LDS copies,
// one workgroup barrier per tile. Q,K: [B,H,S,Dh]; Vt: [B,H,Dh,S];
// ctx out: [B,S,H*Dh] bf16.
// ---------------------------------------------------------------------------
__launch_bounds__(256)
__global__ void attn_kernel(const bf16* __restrict__ Q,
                            const bf16* __restrict__ Km,
                            const bf16* __restrict__ Vt,
                            bf16* __restrict__ ctx)
{
  constexpr int S = 2048, DH = 64;
  constexpr int KST = 72;           // padded K-tile row stride (elements)
  constexpr int VST = 40;           // padded V-tile row stride (elements)
  __shared__ bf16 kbuf[2][32 * KST];   // kv-major 32 x 64 tile
  __shared__ bf16 vbuf[2][DH * VST];   // d-major  64 x 32 tile
  __shared__ bf16 pbuf[8][16 * 40];    // per-wave 16 x 32 P tile

  const int tid  = threadIdx.x;
  const int lane = tid & 31;
  const int w    = tid >> 5;
  const int hf   = lane >> 4;
  const int ln16 = lane & 15;
  const int h = blockIdx.y, b = blockIdx.z;
  const size_t bh = (size_t)(b * 4 + h);

  const bf16* Qp = Q  + bh * S * DH;
  const bf16* Kp = Km + bh * S * DH;
  const bf16* Vp = Vt + bh * DH * S;
  const int q0row = blockIdx.x * 128 + w * 16;

  // staging: each thread moves one 16B chunk of the K tile and one of V.
  const int kidx = tid * 8;                 // element in 32x64 K tile
  const int krow = kidx >> 6, kcol = kidx & 63;
  const int vrow = kidx >> 5, vcol = kidx & 31;  // element in 64x32 V tile

  auto stage = [&](int j0, int buf) {
    stage16(Kp + (size_t)(j0 + krow) * DH + kcol, &kbuf[buf][krow * KST + kcol]);
    stage16(Vp + (size_t)vrow * S + j0 + vcol,    &vbuf[buf][vrow * VST + vcol]);
  };

  // Q A-fragments: dh 0..31 and 32..63.
  v16bf aq[2];
  {
    const bf16* qp = Qp + (size_t)(q0row + ln16) * DH + hf * 8;
    aq[0] = load_frag(qp,      qp + 16);
    aq[1] = load_frag(qp + 32, qp + 48);
  }

  v8f m_run = {}, l_run = {};
  v8f cacc[4];
#pragma unroll
  for (int t = 0; t < 4; ++t) { v8f z = {}; cacc[t] = z; }
#pragma unroll
  for (int e = 0; e < 8; ++e) { m_run[e] = -1e30f; l_run[e] = 0.0f; }

  const float scale = 0.125f;   // Dh^-0.5

  stage(0, 0);
  for (int j0 = 0; j0 < S; j0 += 32) {
    const int buf = (j0 >> 5) & 1;
    async_join();            // own async copies done
    __syncthreads();         // everyone's copies done / prior reads done
    if (j0 + 32 < S) stage(j0 + 32, buf ^ 1);

    const bf16* kb = kbuf[buf];
    const bf16* vb = vbuf[buf];

    // ---- S tile = Q * K^T ----
    v8f s0 = {}, s1 = {};
#pragma unroll
    for (int kk = 0; kk < 2; ++kk) {
      const bf16* kp0 = kb + (size_t)ln16 * KST + kk * 32 + hf * 8;
      v16bf bk0 = load_frag(kp0, kp0 + 16);
      const bf16* kp1 = kb + (size_t)(16 + ln16) * KST + kk * 32 + hf * 8;
      v16bf bk1 = load_frag(kp1, kp1 + 16);
      s0 = __builtin_amdgcn_wmma_f32_16x16x32_bf16(false, aq[kk], false, bk0,
                                                   (short)0, s0, false, false);
      s1 = __builtin_amdgcn_wmma_f32_16x16x32_bf16(false, aq[kk], false, bk1,
                                                   (short)0, s1, false, false);
    }

    // ---- online softmax (row stats in C-fragment layout) ----
    v8f mt;
#pragma unroll
    for (int e = 0; e < 8; ++e) {
      s0[e] *= scale; s1[e] *= scale;
      mt[e] = fmaxf(s0[e], s1[e]);
    }
#pragma unroll
    for (int off = 1; off < 16; off <<= 1)
#pragma unroll
      for (int e = 0; e < 8; ++e)
        mt[e] = fmaxf(mt[e], __shfl_xor(mt[e], off, 32));

    v8f alpha, rs;
#pragma unroll
    for (int e = 0; e < 8; ++e) {
      float mn = fmaxf(m_run[e], mt[e]);
      alpha[e] = __expf(m_run[e] - mn);
      m_run[e] = mn;
      s0[e] = __expf(s0[e] - mn);
      s1[e] = __expf(s1[e] - mn);
      rs[e] = s0[e] + s1[e];
    }
#pragma unroll
    for (int off = 1; off < 16; off <<= 1)
#pragma unroll
      for (int e = 0; e < 8; ++e)
        rs[e] += __shfl_xor(rs[e], off, 32);
#pragma unroll
    for (int e = 0; e < 8; ++e) {
      l_run[e] = l_run[e] * alpha[e] + rs[e];
      cacc[0][e] *= alpha[e]; cacc[1][e] *= alpha[e];
      cacc[2][e] *= alpha[e]; cacc[3][e] *= alpha[e];
    }

    // ---- C-layout P -> LDS -> A-fragment (wave-private) ----
    bf16* pb = pbuf[w];
#pragma unroll
    for (int e = 0; e < 8; ++e) {
      int pr = e + 8 * hf;
      pb[pr * 40 + ln16]      = (bf16)s0[e];
      pb[pr * 40 + 16 + ln16] = (bf16)s1[e];
    }
    const bf16* pp = pb + ln16 * 40 + hf * 8;
    v16bf pf = load_frag(pp, pp + 16);

    // ---- ctx += P * V ----
#pragma unroll
    for (int t = 0; t < 4; ++t) {
      const bf16* vp = vb + (size_t)(t * 16 + ln16) * VST + hf * 8;
      v16bf bv = load_frag(vp, vp + 16);
      cacc[t] = __builtin_amdgcn_wmma_f32_16x16x32_bf16(false, pf, false, bv,
                                                        (short)0, cacc[t], false, false);
    }
  }

  // ---- normalize and write ctx [B,S,H*Dh] ----
#pragma unroll
  for (int t = 0; t < 4; ++t)
#pragma unroll
    for (int e = 0; e < 8; ++e) {
      int qr = q0row + e + 8 * hf;
      int d  = t * 16 + ln16;
      float v = cacc[t][e] / l_run[e];
      ctx[((size_t)b * S + qr) * 256 + h * 64 + d] = (bf16)v;
    }
}

// ---------------------------------------------------------------------------
// Elementwise helpers
// ---------------------------------------------------------------------------
__global__ void cvt_kernel(const float* __restrict__ s, bf16* __restrict__ d, int n) {
  int i = blockIdx.x * blockDim.x + threadIdx.x;
  if (i < n) d[i] = (bf16)s[i];
}

__global__ void pack_desc_kernel(const float* __restrict__ desc, bf16* __restrict__ xcat) {
  int i = blockIdx.x * blockDim.x + threadIdx.x;   // M*256 threads
  int m = i >> 8, c = i & 255;
  xcat[(size_t)m * 512 + c] = (bf16)desc[(size_t)m * 256 + c];
}

__global__ void rotary_kernel(const bf16* __restrict__ qkv, const float* __restrict__ kp,
                              bf16* __restrict__ Q, bf16* __restrict__ K,
                              bf16* __restrict__ Vt)
{
  int i = blockIdx.x * blockDim.x + threadIdx.x;   // B*H*S*32 threads (dh pairs)
  int p = i & 31;
  int s = (i >> 5) & 2047;
  int h = (i >> 16) & 3;
  int b = i >> 18;
  size_t qi = ((size_t)(b * 2048 + s)) * 768 + h * 192 + (p * 2) * 3;
  float q0 = (float)qkv[qi + 0], k0 = (float)qkv[qi + 1], v0 = (float)qkv[qi + 2];
  float q1 = (float)qkv[qi + 3], k1 = (float)qkv[qi + 4], v1 = (float)qkv[qi + 5];
  size_t cb = ((size_t)(b * 2048 + s)) * 64 + p * 2;
  const float* cosp = kp;
  const float* sinp = kp + (size_t)8 * 2048 * 64;
  float c0 = cosp[cb], c1 = cosp[cb + 1], sn0 = sinp[cb], sn1 = sinp[cb + 1];
  float qo0 = q0 * c0 - q1 * sn0, qo1 = q1 * c1 + q0 * sn1;
  float ko0 = k0 * c0 - k1 * sn0, ko1 = k1 * c1 + k0 * sn1;
  size_t o = ((size_t)(b * 4 + h) * 2048 + s) * 64 + p * 2;
  Q[o] = (bf16)qo0; Q[o + 1] = (bf16)qo1;
  K[o] = (bf16)ko0; K[o + 1] = (bf16)ko1;
  size_t vo = ((size_t)(b * 4 + h) * 64 + p * 2) * 2048 + s;
  Vt[vo] = (bf16)v0; Vt[vo + 2048] = (bf16)v1;
}

__launch_bounds__(256)
__global__ void ln_gelu_kernel(const float* __restrict__ y, const float* __restrict__ g,
                               const float* __restrict__ bb, bf16* __restrict__ h)
{
  __shared__ float red[256];
  const int t = threadIdx.x;
  const size_t row = blockIdx.x;
  float x0 = y[row * 512 + t];
  float x1 = y[row * 512 + 256 + t];
  red[t] = x0 + x1;
  __syncthreads();
  for (int st = 128; st > 0; st >>= 1) { if (t < st) red[t] += red[t + st]; __syncthreads(); }
  float mean = red[0] * (1.0f / 512.0f);
  __syncthreads();
  red[t] = x0 * x0 + x1 * x1;
  __syncthreads();
  for (int st = 128; st > 0; st >>= 1) { if (t < st) red[t] += red[t + st]; __syncthreads(); }
  float var  = red[0] * (1.0f / 512.0f) - mean * mean;
  float rstd = rsqrtf(var + 1e-5f);
  float n0 = (x0 - mean) * rstd * g[t] + bb[t];
  float n1 = (x1 - mean) * rstd * g[t + 256] + bb[t + 256];
  n0 = 0.5f * n0 * (1.0f + erff(n0 * 0.70710678118f));
  n1 = 0.5f * n1 * (1.0f + erff(n1 * 0.70710678118f));
  h[row * 512 + t]       = (bf16)n0;
  h[row * 512 + 256 + t] = (bf16)n1;
}

// ---------------------------------------------------------------------------
extern "C" void kernel_launch(void* const* d_in, const int* in_sizes, int n_in,
                              void* d_out, int out_size, void* d_ws, size_t ws_size,
                              hipStream_t stream)
{
  const float* desc   = (const float*)d_in[0];
  const float* keyp   = (const float*)d_in[1];
  const float* Wqkv_w = (const float*)d_in[2];
  const float* Wqkv_b = (const float*)d_in[3];
  const float* Wo_w   = (const float*)d_in[4];
  const float* Wo_b   = (const float*)d_in[5];
  const float* W1_w   = (const float*)d_in[6];
  const float* W1_b   = (const float*)d_in[7];
  const float* ln_g   = (const float*)d_in[8];
  const float* ln_b   = (const float*)d_in[9];
  const float* W2_w   = (const float*)d_in[10];
  const float* W2_b   = (const float*)d_in[11];
  (void)in_sizes; (void)n_in; (void)out_size; (void)ws_size;

  constexpr int M = 8 * 2048;

  char* ws = (char*)d_ws;
  size_t off = 0;
  auto take = [&](size_t bytes) -> char* {
    char* p = ws + off;
    off += (bytes + 255) & ~(size_t)255;
    return p;
  };
  bf16*  xcat = (bf16*)take((size_t)M * 512 * 2);
  bf16*  qkv  = (bf16*)take((size_t)M * 768 * 2);
  bf16*  Qb   = (bf16*)take((size_t)M * 256 * 2);
  bf16*  Kb   = (bf16*)take((size_t)M * 256 * 2);
  bf16*  Vtb  = (bf16*)take((size_t)M * 256 * 2);
  bf16*  ctx  = (bf16*)take((size_t)M * 256 * 2);
  float* y1   = (float*)take((size_t)M * 512 * 4);
  bf16*  h1   = (bf16*)take((size_t)M * 512 * 2);
  bf16*  wqkv = (bf16*)take((size_t)768 * 256 * 2);
  bf16*  wo   = (bf16*)take((size_t)256 * 256 * 2);
  bf16*  w1   = (bf16*)take((size_t)512 * 512 * 2);
  bf16*  w2   = (bf16*)take((size_t)256 * 512 * 2);

  cvt_kernel<<<(768 * 256 + 255) / 256, 256, 0, stream>>>(Wqkv_w, wqkv, 768 * 256);
  cvt_kernel<<<(256 * 256 + 255) / 256, 256, 0, stream>>>(Wo_w,   wo,   256 * 256);
  cvt_kernel<<<(512 * 512 + 255) / 256, 256, 0, stream>>>(W1_w,   w1,   512 * 512);
  cvt_kernel<<<(256 * 512 + 255) / 256, 256, 0, stream>>>(W2_w,   w2,   256 * 512);
  pack_desc_kernel<<<(M * 256) / 256, 256, 0, stream>>>(desc, xcat);

  gemm_bf16_kernel<0><<<dim3(M / 128, 768 / 64), 256, 0, stream>>>(
      xcat, 512, wqkv, 256, Wqkv_b, qkv, 768, 0, nullptr, 0);

  rotary_kernel<<<(8 * 4 * 2048 * 32) / 256, 256, 0, stream>>>(qkv, keyp, Qb, Kb, Vtb);

  attn_kernel<<<dim3(2048 / 128, 4, 8), 256, 0, stream>>>(Qb, Kb, Vtb, ctx);

  gemm_bf16_kernel<0><<<dim3(M / 128, 256 / 64), 256, 0, stream>>>(
      ctx, 256, wo, 256, Wo_b, xcat, 512, 256, nullptr, 0);

  gemm_bf16_kernel<1><<<dim3(M / 128, 512 / 64), 256, 0, stream>>>(
      xcat, 512, w1, 512, W1_b, y1, 512, 0, nullptr, 0);

  ln_gelu_kernel<<<M, 256, 0, stream>>>(y1, ln_g, ln_b, h1);

  gemm_bf16_kernel<2><<<dim3(M / 128, 256 / 64), 256, 0, stream>>>(
      h1, 512, w2, 512, W2_b, d_out, 256, 0, desc, 256);
}
